// PyTorchBCRF_43671227465728
// MI455X (gfx1250) — compile-verified, matrix operand order checked by reading, not run on hardware
//
#include <hip/hip_runtime.h>
#include <hip/hip_bf16.h>

#define HH 480
#define WW 640
#define HW (HH * WW)
#define NSEM 12
#define NINS 16
#define NCH  (NSEM + NINS)   // 28
#define PADC 16              // padded sem channel count inside ws buffers
#define RB   2               // bilateral radius
#define INV2A2 (1.0f / 18.0f)       // 1/(2*3^2)
#define INV2B2 (1.0f / 0.045f)      // 1/(2*0.15^2)

typedef __attribute__((ext_vector_type(16))) _Float16 v16h;
typedef __attribute__((ext_vector_type(8)))  float    v8f;

// channel remap inside padded 32-row buffers: sem 0..11 -> rows 0..11,
// ins 0..15 -> rows 16..31
__device__ __forceinline__ constexpr int chmap(int c) {
    return (c < NSEM) ? c : (c + (PADC - NSEM));
}

// Gaussian kernel, sigma=3, r=6 (precomputed, normalized)
__device__ __constant__ float c_gk[13] = {
    0.0185441f, 0.0341670f, 0.0563318f, 0.0831086f, 0.1097193f, 0.1296180f,
    0.1370229f,
    0.1296180f, 0.1097193f, 0.0831086f, 0.0563318f, 0.0341670f, 0.0185441f
};

__device__ __forceinline__ int clampi(int v, int lo, int hi) {
    return v < lo ? lo : (v > hi ? hi : v);
}

// ---------------------------------------------------------------------------
// Precompute normalized bilateral weights (image is constant across iters)
// wn[t*HW + p] = w_t(p) / sum_t w_t(p)
// ---------------------------------------------------------------------------
__global__ __launch_bounds__(256) void wnorm_kernel(
    const float* __restrict__ img, float* __restrict__ wn) {
    int p = blockIdx.x * 256 + threadIdx.x;
    if (p >= HW) return;
    int y = p / WW, x = p - y * WW;
    float i0 = img[p], i1 = img[HW + p], i2 = img[2 * HW + p];
    float w[25];
    float den = 0.f;
    #pragma unroll
    for (int dy = 0; dy < 5; ++dy) {
        int yy = clampi(y + dy - RB, 0, HH - 1);
        #pragma unroll
        for (int dx = 0; dx < 5; ++dx) {
            int xx = clampi(x + dx - RB, 0, WW - 1);
            int q = yy * WW + xx;
            float d0 = img[q] - i0;
            float d1 = img[HW + q] - i1;
            float d2 = img[2 * HW + q] - i2;
            float cd = d0 * d0 + d1 * d1 + d2 * d2;
            float sd = (float)((dy - RB) * (dy - RB) + (dx - RB) * (dx - RB));
            float wv = __expf(-(sd * INV2A2 + cd * INV2B2));
            w[dy * 5 + dx] = wv;
            den += wv;
        }
    }
    float r = 1.0f / den;
    #pragma unroll
    for (int t = 0; t < 25; ++t) wn[t * HW + p] = w[t] * r;
}

// ---------------------------------------------------------------------------
// Gather per-instance-channel params + cross matrices (tiny, one block)
// ---------------------------------------------------------------------------
__global__ void init_small_kernel(
    const int* __restrict__ labels,
    const float* __restrict__ cis,   // cross_ins_sem  (9 x 12)
    const float* __restrict__ csi,   // cross_sem_ins  (9 x 12)
    const float* __restrict__ isw, const float* __restrict__ ibw,
    float* __restrict__ Mis,   // 16 x 12  : Mis[i*12+o] = cis[labels[i]][o]
    float* __restrict__ Msi,   // 12 x 16  : Msi[i*16+o] = csi[labels[o]][i]
    float* __restrict__ swi, float* __restrict__ bwi) {
    int t = threadIdx.x;
    if (t < NINS) {
        int l = labels[t];
        swi[t] = isw[l];
        bwi[t] = ibw[l];
    }
    if (t < NINS * NSEM) {
        int i = t / NSEM, o = t - i * NSEM;
        Mis[t] = cis[labels[i] * NSEM + o];
        int i2 = t / NINS, o2 = t - i2 * NINS;
        Msi[t] = csi[labels[o2] * NSEM + i2];
    }
}

// ---------------------------------------------------------------------------
// Dual softmax over channels (sem 12ch + ins 16ch), per pixel, stable.
// Safe in-place (each thread owns its pixel's channel column).
// ---------------------------------------------------------------------------
__global__ __launch_bounds__(256) void softmax2_kernel(
    const float* __restrict__ semIn, const float* __restrict__ insIn,
    float* __restrict__ semOut, float* __restrict__ insOut) {
    int p = blockIdx.x * 256 + threadIdx.x;
    if (p >= HW) return;
    {
        float v[NSEM], m = -3.0e38f;
        #pragma unroll
        for (int c = 0; c < NSEM; ++c) { v[c] = semIn[c * HW + p]; m = fmaxf(m, v[c]); }
        float s = 0.f;
        #pragma unroll
        for (int c = 0; c < NSEM; ++c) { v[c] = __expf(v[c] - m); s += v[c]; }
        float r = 1.0f / s;
        #pragma unroll
        for (int c = 0; c < NSEM; ++c) semOut[c * HW + p] = v[c] * r;
    }
    {
        float v[NINS], m = -3.0e38f;
        #pragma unroll
        for (int c = 0; c < NINS; ++c) { v[c] = insIn[c * HW + p]; m = fmaxf(m, v[c]); }
        float s = 0.f;
        #pragma unroll
        for (int c = 0; c < NINS; ++c) { v[c] = __expf(v[c] - m); s += v[c]; }
        float r = 1.0f / s;
        #pragma unroll
        for (int c = 0; c < NINS; ++c) insOut[c * HW + p] = v[c] * r;
    }
}

// ---------------------------------------------------------------------------
// Separable Gaussian blur, zero-padded (matches conv padding). One logical
// channel per blockIdx.y (0..27); rows remapped into padded 32-row buffers.
// ---------------------------------------------------------------------------
__global__ __launch_bounds__(256) void blurW_kernel(
    const float* __restrict__ in, float* __restrict__ out) {
    int cc = (int)blockIdx.y;
    int c  = (cc < NSEM) ? cc : (cc + (PADC - NSEM));   // padded row
    int p = blockIdx.x * 256 + threadIdx.x;
    if (p >= HW) return;
    int y = p / WW, x = p - y * WW;
    const float* src = in + (size_t)c * HW + y * WW;
    float acc = 0.f;
    #pragma unroll
    for (int t = -6; t <= 6; ++t) {
        int xx = x + t;
        float v = (xx >= 0 && xx < WW) ? src[clampi(xx, 0, WW - 1)] : 0.f;
        acc += c_gk[t + 6] * v;
    }
    out[(size_t)c * HW + p] = acc;
}

__global__ __launch_bounds__(256) void blurH_kernel(
    const float* __restrict__ in, float* __restrict__ out) {
    int cc = (int)blockIdx.y;
    int c  = (cc < NSEM) ? cc : (cc + (PADC - NSEM));   // padded row
    int p = blockIdx.x * 256 + threadIdx.x;
    if (p >= HW) return;
    int y = p / WW, x = p - y * WW;
    const float* src = in + (size_t)c * HW;
    float acc = 0.f;
    #pragma unroll
    for (int t = -6; t <= 6; ++t) {
        int yy = y + t;
        float v = (yy >= 0 && yy < HH) ? src[clampi(yy, 0, HH - 1) * WW + x] : 0.f;
        acc += c_gk[t + 6] * v;
    }
    out[(size_t)c * HW + p] = acc;
}

// ---------------------------------------------------------------------------
// Bilateral apply: all 28 logical channels per thread, precomputed weights
// (one wn load per tap, 28 FMAs).  Edge-clamped.  Padded row remap folds
// to constant offsets under full unroll.
// ---------------------------------------------------------------------------
__global__ __launch_bounds__(256) void bilateral_kernel(
    const float* __restrict__ q, const float* __restrict__ wn,
    float* __restrict__ out) {
    int p = blockIdx.x * 256 + threadIdx.x;
    if (p >= HW) return;
    int y = p / WW, x = p - y * WW;
    float acc[NCH];
    #pragma unroll
    for (int c = 0; c < NCH; ++c) acc[c] = 0.f;
    for (int dy = 0; dy < 5; ++dy) {
        int yy = clampi(y + dy - RB, 0, HH - 1);
        for (int dx = 0; dx < 5; ++dx) {
            int xx = clampi(x + dx - RB, 0, WW - 1);
            int qi = yy * WW + xx;
            float w = wn[(dy * 5 + dx) * HW + p];
            #pragma unroll
            for (int c = 0; c < NCH; ++c) acc[c] += w * q[chmap(c) * HW + qi];
        }
    }
    #pragma unroll
    for (int c = 0; c < NCH; ++c) out[chmap(c) * HW + p] = acc[c];
}

// ---------------------------------------------------------------------------
// WMMA channel-mix (fully specialized at compile time):
//   out(16pad x 16px) = M^T(OxI) * B(I x 16px) + Cbase
//   B[i][px] = wA[i]*inA[i][px] + wB[i]*inB[i][px]   (weights/inB optional)
//   IDENT -> identity mixing; CPAD -> Cbase has 16 valid (zero-padded) rows.
// A rows o>=O and C rows o2>=O are zeroed, so D rows O..15 are exactly 0 and
// the 16-row store into the padded out-buffer is fully unconditional.
// One wave builds A once, then runs TPW tiles (4 WMMAs) reusing it.
// ---------------------------------------------------------------------------
template <int I, int O, bool HASB, bool HASW, bool IDENT, bool CPAD, int TPW>
__global__ __launch_bounds__(256) void wmma_compat_t(
    const float* __restrict__ Cbase,
    const float* __restrict__ inA, const float* __restrict__ inB,
    const float* __restrict__ wA, const float* __restrict__ wB,
    const float* __restrict__ Mmat,
    float* __restrict__ out) {
    const int wave  = threadIdx.x >> 5;
    const int lane  = threadIdx.x & 31;
    const int half  = lane >> 4;
    const int o     = lane & 15;
    const int tile0 = (blockIdx.x * (int)(blockDim.x >> 5) + wave) * TPW;

    // Build A (= M^T, f16, K padded to 32) once; reused by all TPW tiles.
    v16h a{};
    #pragma unroll
    for (int e = 0; e < 16; ++e) {
        int K = (e < 8) ? (half * 8 + e) : (16 + half * 8 + (e - 8));
        float av;
        if (IDENT) {
            av = (K == o) ? 1.f : 0.f;          // I == O == 16
        } else {
            int Kc = (K < I) ? K : (I - 1);
            int oc = (o < O) ? o : (O - 1);
            av = Mmat[Kc * O + oc];
            if (K >= I || o >= O) av = 0.f;     // branchless mask
        }
        a[e] = (_Float16)av;
    }

    #pragma unroll
    for (int t = 0; t < TPW; ++t) {
        const int px = (tile0 + t) * 16 + o;    // HW % (16*8*TPW) == 0

        v16h b{};
        #pragma unroll
        for (int e = 0; e < 16; ++e) {
            int K  = (e < 8) ? (half * 8 + e) : (16 + half * 8 + (e - 8));
            int Kc = (K < I) ? K : (I - 1);     // clamped, unconditional load
            float bv = inA[Kc * HW + px];
            if (HASW) bv *= wA[Kc];
            if (HASB) {
                float u = inB[Kc * HW + px];
                if (HASW) u *= wB[Kc];
                bv += u;
            }
            if (K >= I) bv = 0.f;               // branchless mask
            b[e] = (_Float16)bv;
        }

        v8f c;
        #pragma unroll
        for (int v = 0; v < 8; ++v) {
            int o2 = v + 8 * half;
            if (CPAD || O == 16) {
                c[v] = Cbase[o2 * HW + px];     // padded: unconditional
            } else {
                int o2c = (o2 < O) ? o2 : (O - 1);
                float cv = Cbase[o2c * HW + px];
                if (o2 >= O) cv = 0.f;          // branchless mask
                c[v] = cv;
            }
        }

        v8f d = __builtin_amdgcn_wmma_f32_16x16x32_f16(
            false, a, false, b, (short)0, c, false, false);

        #pragma unroll
        for (int v = 0; v < 8; ++v) {           // rows O..15 are zeros
            int o2 = v + 8 * half;
            out[o2 * HW + px] = d[v];           // padded out: unconditional
        }
    }
}

// ---------------------------------------------------------------------------
extern "C" void kernel_launch(void* const* d_in, const int* in_sizes, int n_in,
                              void* d_out, int out_size, void* d_ws, size_t ws_size,
                              hipStream_t stream) {
    (void)in_sizes; (void)n_in; (void)out_size; (void)ws_size;
    const float* image   = (const float*)d_in[0];   // 3 x HW
    const float* semL    = (const float*)d_in[1];   // 12 x HW
    const float* insL    = (const float*)d_in[2];   // 16 x HW
    const int*   labels  = (const int*)  d_in[3];   // 16
    const float* semSW   = (const float*)d_in[4];   // 12
    const float* semBW   = (const float*)d_in[5];   // 12
    const float* semM    = (const float*)d_in[6];   // 12 x 12 (i,o)
    const float* insSW   = (const float*)d_in[7];   // 9
    const float* insBW   = (const float*)d_in[8];   // 9
    const float* crossIS = (const float*)d_in[9];   // 9 x 12
    const float* crossSI = (const float*)d_in[10];  // 9 x 12

    // Padded buffers: 32 rows each (sem rows 0..15 [12 used], ins rows 16..31)
    float* ws   = (float*)d_ws;
    float* wn   = ws;                         // 25*HW  bilateral weights
    float* qbuf = wn   + (size_t)25 * HW;     // 32*HW  q / new logits
    float* t1   = qbuf + (size_t)32 * HW;     // 32*HW  blur temp, then bilateral
    float* blur = t1   + (size_t)32 * HW;     // 32*HW  blur out, then softmax(tmp)
    float* tmp  = blur + (size_t)32 * HW;     // 32*HW  tmp_sem / tmp_ins
    float* Mis  = tmp  + (size_t)32 * HW;     // 192
    float* Msi  = Mis + NINS * NSEM;          // 192
    float* swi  = Msi + NSEM * NINS;          // 16
    float* bwi  = swi + NINS;                 // 16

    float* qSem = qbuf;              float* qIns = qbuf + (size_t)PADC * HW;
    float* bSem = blur;              float* bIns = blur + (size_t)PADC * HW;
    float* lSem = t1;                float* lIns = t1   + (size_t)PADC * HW;
    float* tSem = tmp;               float* tIns = tmp  + (size_t)PADC * HW;

    const int PB = HW / 256;               // 1200 pixel blocks
    const int WB = (HW / 16) / (8 * 4);    // 600 wmma blocks (8 waves x 4 tiles)

    wnorm_kernel<<<PB, 256, 0, stream>>>(image, wn);
    init_small_kernel<<<1, 192, 0, stream>>>(labels, crossIS, crossSI,
                                             insSW, insBW, Mis, Msi, swi, bwi);

    const float* curSem = semL;
    const float* curIns = insL;
    for (int it = 0; it < 3; ++it) {
        // q = softmax(cur)
        softmax2_kernel<<<PB, 256, 0, stream>>>(curSem, curIns, qSem, qIns);
        // separable Gaussian blur (zero pad):  q -> t1 -> blur
        blurW_kernel<<<dim3(PB, NCH), 256, 0, stream>>>(qbuf, t1);
        blurH_kernel<<<dim3(PB, NCH), 256, 0, stream>>>(t1, blur);
        // bilateral (edge pad, precomputed weights):  q -> t1
        bilateral_kernel<<<PB, 256, 0, stream>>>(qbuf, wn, t1);
        // tmp_sem = sem_logits + semM^T * (sw.*blur + bw.*bil)   [WMMA]
        wmma_compat_t<NSEM, NSEM, true, true, false, false, 4>
            <<<WB, 256, 0, stream>>>(semL, bSem, lSem, semSW, semBW, semM, tSem);
        // tmp_ins = ins_logits + I * (swi.*blur + bwi.*bil)      [WMMA, identity]
        wmma_compat_t<NINS, NINS, true, true, true, false, 4>
            <<<WB, 256, 0, stream>>>(insL, bIns, lIns, swi, bwi, nullptr, tIns);
        // s = softmax(tmp)  (into blur buffer, now free)
        softmax2_kernel<<<PB, 256, 0, stream>>>(tSem, tIns, bSem, bIns);
        // new_sem = tmp_sem + Mis^T * s_ins   [WMMA; Cbase=tSem is padded]
        wmma_compat_t<NINS, NSEM, false, false, false, true, 4>
            <<<WB, 256, 0, stream>>>(tSem, bIns, nullptr, nullptr, nullptr, Mis, qSem);
        // new_ins = tmp_ins + Msi^T * s_sem   [WMMA]
        wmma_compat_t<NSEM, NINS, false, false, false, false, 4>
            <<<WB, 256, 0, stream>>>(tIns, bSem, nullptr, nullptr, nullptr, Msi, qIns);
        curSem = qSem;
        curIns = qIns;
    }
    // final outputs: softmax of last-iteration logits (compact d_out layout)
    float* outSem = (float*)d_out;
    float* outIns = outSem + (size_t)NSEM * HW;
    softmax2_kernel<<<PB, 256, 0, stream>>>(qSem, qIns, outSem, outIns);
}
